// MaceModel_27788438405222
// MI455X (gfx1250) — matrix operand order repaired and checked by reading, not compiled, not run on hardware
//
#include <hip/hip_runtime.h>
#include <hip/hip_bf16.h>

// ---------------------------------------------------------------------------
// MACE-like GNN, dead-code-eliminated (only l=0 channel is live; sh[:,0]==1).
// Dominant GEMM (rad[E,64] @ w_r2[li][:, :32]) runs on v_wmma_f32_16x16x32_f16.
// ---------------------------------------------------------------------------

#define NN      20000
#define EE      320000
#define CC      32
#define SS      64
#define GG      64
#define NRAD    8
#define HRAD    64
#define OUTS    16
#define HHEAD   64
#define NL      2
#define RCUT    5.0f

typedef __attribute__((ext_vector_type(16))) _Float16 v16h;
typedef __attribute__((ext_vector_type(8)))  float    v8f;

// ---------------------------------------------------------------------------
// Kernel 0: convert live slice of w_r2 (NL,64,96) -> fp16 (NL,64,32), cols 0..31
// ---------------------------------------------------------------------------
__global__ void k_convert_wr2(const float* __restrict__ w_r2, _Float16* __restrict__ wr2h) {
    int idx = blockIdx.x * blockDim.x + threadIdx.x;           // [0, NL*64*32)
    if (idx >= NL * HRAD * CC) return;
    int c  = idx % CC;
    int k  = (idx / CC) % HRAD;
    int li = idx / (CC * HRAD);
    wr2h[idx] = (_Float16)w_r2[(li * HRAD + k) * (3 * CC) + c];
}

// ---------------------------------------------------------------------------
// Kernel 1: per-edge radial basis -> rad = silu(bessel @ w_r1 + b_r1), fp16 out
// ---------------------------------------------------------------------------
__global__ void k_radial(const float* __restrict__ vectors,
                         const float* __restrict__ w_r1, const float* __restrict__ b_r1,
                         _Float16* __restrict__ radh) {
    __shared__ float s_w[NRAD * HRAD];
    __shared__ float s_b[HRAD];
    for (int i = threadIdx.x; i < NRAD * HRAD; i += blockDim.x) s_w[i] = w_r1[i];
    for (int i = threadIdx.x; i < HRAD; i += blockDim.x) s_b[i] = b_r1[i];
    __syncthreads();

    int e = blockIdx.x * blockDim.x + threadIdx.x;
    if (e >= EE) return;

    float vx = vectors[e * 3 + 0], vy = vectors[e * 3 + 1], vz = vectors[e * 3 + 2];
    float r2 = vx * vx + vy * vy + vz * vz;
    float r  = (r2 == 0.0f) ? 0.0f : sqrtf(r2);
    float rs = fmaxf(r, 1e-6f);
    float u  = fminf(fmaxf(r / RCUT, 0.0f), 1.0f);
    float env = (1.0f - u) * (1.0f - u) * (1.0f + 2.0f * u);
    const float pref = 0.63245553203367588f;                   // sqrt(2/RCUT)
    float bes[NRAD];
#pragma unroll
    for (int k = 0; k < NRAD; ++k) {
        float n = (float)(k + 1);
        bes[k] = pref * __sinf(n * 3.14159265358979323846f * rs / RCUT) / rs * env;
    }
#pragma unroll 4
    for (int j = 0; j < HRAD; ++j) {
        float acc = s_b[j];
#pragma unroll
        for (int k = 0; k < NRAD; ++k) acc = fmaf(bes[k], s_w[k * HRAD + j], acc);
        float sil = acc / (1.0f + __expf(-acc));               // silu
        radh[e * HRAD + j] = (_Float16)sil;
    }
}

// ---------------------------------------------------------------------------
// Kernel 2: h init from species embedding
// ---------------------------------------------------------------------------
__global__ void k_init_h(const int* __restrict__ species, const float* __restrict__ emb,
                         float* __restrict__ h) {
    int idx = blockIdx.x * blockDim.x + threadIdx.x;
    if (idx >= NN * CC) return;
    int n = idx / CC, c = idx % CC;
    h[idx] = emb[species[n] * CC + c];
}

// ---------------------------------------------------------------------------
// Kernel 3 (per layer): WMMA edge GEMM + elementwise * h[sender] + scatter-add.
//   Tile = 16 edges. D(16x16,f32) = A(16x32,f16) x B(32x16,f16), K=64 in 2 steps,
//   N=32 in 2 halves -> 4 x v_wmma_f32_16x16x32_f16 per tile.
// CDNA5 wave32 fragment layouts (ISA 7.12.2):
//   A: lane L -> row M=L&15;  halves: lanes 0-15 hold K 0-7,16-23; 16-31 hold 8-15,24-31
//   B: lane L -> col N=L&15;  same K split as A: lanes 0-15 hold K 0-7,16-23; 16-31 hold 8-15,24-31
//   D: lane L -> col N=L&15;  d[i] = row (L>>4)*8 + i
// ---------------------------------------------------------------------------
__global__ void k_edge_scatter(const _Float16* __restrict__ radh,
                               const _Float16* __restrict__ wr2h_l,   // 64x32 fp16
                               const float* __restrict__ h_in,
                               const int* __restrict__ senders,
                               const int* __restrict__ receivers,
                               float* __restrict__ agg,
                               int numTiles) {
    const int lane    = threadIdx.x & 31;
    const int waveId  = (blockIdx.x * blockDim.x + threadIdx.x) >> 5;
    const int nWaves  = (gridDim.x * blockDim.x) >> 5;
    const int hi      = lane >> 4;       // lane-half selector
    const int lo      = lane & 15;

    // ---- B fragments: 2 k-tiles x 2 n-halves, loaded once per wave ----
    v16h bfrag[2][2];
#pragma unroll
    for (int kt = 0; kt < 2; ++kt)
#pragma unroll
        for (int nh = 0; nh < 2; ++nh)
#pragma unroll
            for (int i = 0; i < 16; ++i) {
                int v = i >> 1, pos = i & 1;
                int k = kt * 32 + ((v < 4) ? 0 : 16) + hi * 8 + (v & 3) * 2 + pos;
                int n = nh * 16 + lo;
                bfrag[kt][nh][i] = wr2h_l[k * CC + n];
            }

    for (int tile = waveId; tile < numTiles; tile += nWaves) {
        const int ebase = tile * 16;

        if (tile + nWaves < numTiles)    // prefetch next tile's A operand
            __builtin_prefetch(&radh[(size_t)(tile + nWaves) * 16 * HRAD], 0, 0);

        // ---- A fragments (two K tiles of 32) ----
        v16h afrag[2];
#pragma unroll
        for (int kt = 0; kt < 2; ++kt)
#pragma unroll
            for (int i = 0; i < 16; ++i) {
                int v = i >> 1, pos = i & 1;
                int k = kt * 32 + ((v < 4) ? 0 : 16) + hi * 8 + (v & 3) * 2 + pos;
                afrag[kt][i] = radh[(size_t)(ebase + lo) * HRAD + k];
            }

        // ---- 4x WMMA, f32 accumulate ----
        v8f c0 = {}, c1 = {};
        c0 = __builtin_amdgcn_wmma_f32_16x16x32_f16(false, afrag[0], false, bfrag[0][0],
                                                    (short)0, c0, false, false);
        c0 = __builtin_amdgcn_wmma_f32_16x16x32_f16(false, afrag[1], false, bfrag[1][0],
                                                    (short)0, c0, false, false);
        c1 = __builtin_amdgcn_wmma_f32_16x16x32_f16(false, afrag[0], false, bfrag[0][1],
                                                    (short)0, c1, false, false);
        c1 = __builtin_amdgcn_wmma_f32_16x16x32_f16(false, afrag[1], false, bfrag[1][1],
                                                    (short)0, c1, false, false);

        // ---- multiply by h[sender] and scatter-add into agg[receiver] ----
        int snd[8], rcv[8];
#pragma unroll
        for (int i = 0; i < 8; ++i) {
            int e = ebase + hi * 8 + i;
            snd[i] = senders[e];
            rcv[i] = receivers[e];
        }
        int c_lo = lo, c_hi = lo + 16;
#pragma unroll
        for (int i = 0; i < 8; ++i) {
            float v0 = c0[i] * h_in[snd[i] * CC + c_lo];
            float v1 = c1[i] * h_in[snd[i] * CC + c_hi];
            atomicAdd(&agg[rcv[i] * CC + c_lo], v0);
            atomicAdd(&agg[rcv[i] * CC + c_hi], v1);
        }
    }
}

// ---------------------------------------------------------------------------
// Kernel 4 (per layer): h_out = agg @ w_mix[li][0] + h_in @ w_sc[li][sp] + h_in
// ---------------------------------------------------------------------------
__global__ void k_node_update(const float* __restrict__ agg,
                              const float* __restrict__ h_in,
                              const float* __restrict__ w_mix0,      // 32x32
                              const float* __restrict__ w_sc_l,      // S x 32 x 32
                              const int* __restrict__ species,
                              float* __restrict__ h_out) {
    __shared__ float s_wm[CC * CC];
    for (int i = threadIdx.x; i < CC * CC; i += blockDim.x) s_wm[i] = w_mix0[i];
    __syncthreads();

    int idx = blockIdx.x * blockDim.x + threadIdx.x;
    if (idx >= NN * CC) return;
    int n = idx / CC, c = idx % CC;

    const float* wsc = w_sc_l + (size_t)species[n] * CC * CC;
    float mixed = 0.0f, sc = 0.0f;
#pragma unroll 8
    for (int k = 0; k < CC; ++k) {
        mixed = fmaf(agg[n * CC + k], s_wm[k * CC + c], mixed);
        sc    = fmaf(h_in[n * CC + k], wsc[k * CC + c], sc);
    }
    h_out[idx] = mixed + sc + h_in[idx];
}

// ---------------------------------------------------------------------------
// Kernel 5: per-node readout MLP + layernorm + gelu head, atomic graph reduce
// ---------------------------------------------------------------------------
__global__ void k_readout(const float* __restrict__ h,
                          const float* __restrict__ w_readout,  // 32x16
                          const float* __restrict__ ln_g, const float* __restrict__ ln_b,
                          const float* __restrict__ w_h1,       // 16x64
                          const float* __restrict__ b_h1,       // 64
                          const float* __restrict__ w_h2,       // 64
                          const float* __restrict__ b_h2,       // 1
                          const int* __restrict__ graph_id,
                          float* __restrict__ gsum, float* __restrict__ gcnt) {
    __shared__ float s_wr[CC * OUTS];
    __shared__ float s_g[OUTS], s_be[OUTS];
    __shared__ float s_w1[OUTS * HHEAD];
    __shared__ float s_b1[HHEAD], s_w2[HHEAD];
    __shared__ float s_b2;
    for (int i = threadIdx.x; i < CC * OUTS; i += blockDim.x) s_wr[i] = w_readout[i];
    for (int i = threadIdx.x; i < OUTS; i += blockDim.x) { s_g[i] = ln_g[i]; s_be[i] = ln_b[i]; }
    for (int i = threadIdx.x; i < OUTS * HHEAD; i += blockDim.x) s_w1[i] = w_h1[i];
    for (int i = threadIdx.x; i < HHEAD; i += blockDim.x) { s_b1[i] = b_h1[i]; s_w2[i] = w_h2[i]; }
    if (threadIdx.x == 0) s_b2 = b_h2[0];
    __syncthreads();

    int n = blockIdx.x * blockDim.x + threadIdx.x;
    if (n >= NN) return;

    float hv[CC];
#pragma unroll
    for (int k = 0; k < CC; ++k) hv[k] = h[n * CC + k];

    float ro[OUTS];
    float mu = 0.0f;
#pragma unroll
    for (int j = 0; j < OUTS; ++j) {
        float a = 0.0f;
#pragma unroll
        for (int k = 0; k < CC; ++k) a = fmaf(hv[k], s_wr[k * OUTS + j], a);
        ro[j] = a; mu += a;
    }
    mu *= (1.0f / OUTS);
    float var = 0.0f;
#pragma unroll
    for (int j = 0; j < OUTS; ++j) { float d = ro[j] - mu; var = fmaf(d, d, var); }
    var *= (1.0f / OUTS);
    float inv = __frsqrt_rn(var + 1e-6f);

    float nrm[OUTS];
#pragma unroll
    for (int j = 0; j < OUTS; ++j) nrm[j] = (ro[j] - mu) * inv * s_g[j] + s_be[j];

    float out = s_b2;
    const float kG = 0.7978845608028654f;  // sqrt(2/pi)
#pragma unroll 4
    for (int j2 = 0; j2 < HHEAD; ++j2) {
        float t = s_b1[j2];
#pragma unroll
        for (int j = 0; j < OUTS; ++j) t = fmaf(nrm[j], s_w1[j * HHEAD + j2], t);
        float g = 0.5f * t * (1.0f + tanhf(kG * (t + 0.044715f * t * t * t)));  // gelu (tanh)
        out = fmaf(g, s_w2[j2], out);
    }

    int gid = graph_id[n];
    atomicAdd(&gsum[gid], out);
    atomicAdd(&gcnt[gid], 1.0f);
}

// ---------------------------------------------------------------------------
// Kernel 6: finalize graph means
// ---------------------------------------------------------------------------
__global__ void k_finalize(const float* __restrict__ gsum, const float* __restrict__ gcnt,
                           const float* __restrict__ scale, const float* __restrict__ shift,
                           float* __restrict__ out) {
    int g = blockIdx.x * blockDim.x + threadIdx.x;
    if (g >= GG) return;
    out[g] = gsum[g] / fmaxf(gcnt[g], 1.0f) * scale[0] + shift[0];
}

// ---------------------------------------------------------------------------
extern "C" void kernel_launch(void* const* d_in, const int* in_sizes, int n_in,
                              void* d_out, int out_size, void* d_ws, size_t ws_size,
                              hipStream_t stream) {
    const float* vectors    = (const float*)d_in[0];
    const int*   species    = (const int*)  d_in[1];
    const int*   senders    = (const int*)  d_in[2];
    const int*   receivers  = (const int*)  d_in[3];
    const int*   graph_id   = (const int*)  d_in[4];
    const float* spec_emb   = (const float*)d_in[5];
    const float* w_r1       = (const float*)d_in[6];
    const float* b_r1       = (const float*)d_in[7];
    const float* w_r2       = (const float*)d_in[8];
    const float* w_mix      = (const float*)d_in[9];
    const float* w_sc       = (const float*)d_in[10];
    const float* w_readout  = (const float*)d_in[11];
    const float* ln_g       = (const float*)d_in[12];
    const float* ln_b       = (const float*)d_in[13];
    const float* w_h1       = (const float*)d_in[14];
    const float* b_h1       = (const float*)d_in[15];
    const float* w_h2       = (const float*)d_in[16];
    const float* b_h2       = (const float*)d_in[17];
    const float* scale      = (const float*)d_in[18];
    const float* shift      = (const float*)d_in[19];
    float* out = (float*)d_out;

    // workspace layout
    char* ws = (char*)d_ws;
    _Float16* radh = (_Float16*)ws;                ws += (size_t)EE * HRAD * sizeof(_Float16);
    _Float16* wr2h = (_Float16*)ws;                ws += (size_t)NL * HRAD * CC * sizeof(_Float16);
    float* h_a  = (float*)ws;                      ws += (size_t)NN * CC * sizeof(float);
    float* h_b  = (float*)ws;                      ws += (size_t)NN * CC * sizeof(float);
    float* agg  = (float*)ws;                      ws += (size_t)NN * CC * sizeof(float);
    float* gsum = (float*)ws;                      ws += GG * sizeof(float);
    float* gcnt = (float*)ws;

    // 0) fp16 weight slice
    k_convert_wr2<<<(NL * HRAD * CC + 255) / 256, 256, 0, stream>>>(w_r2, wr2h);
    // 1) radial features (layer-independent)
    k_radial<<<(EE + 255) / 256, 256, 0, stream>>>(vectors, w_r1, b_r1, radh);
    // 2) h init
    k_init_h<<<(NN * CC + 255) / 256, 256, 0, stream>>>(species, spec_emb, h_a);

    const int numTiles = EE / 16;
    float* h_in = h_a;
    float* h_out = h_b;
    for (int li = 0; li < NL; ++li) {
        hipMemsetAsync(agg, 0, (size_t)NN * CC * sizeof(float), stream);
        k_edge_scatter<<<1024, 256, 0, stream>>>(radh, wr2h + (size_t)li * HRAD * CC,
                                                 h_in, senders, receivers, agg, numTiles);
        k_node_update<<<(NN * CC + 255) / 256, 256, 0, stream>>>(
            agg, h_in,
            w_mix + (size_t)li * 3 * CC * CC,          // w_mix[li][0]
            w_sc  + (size_t)li * SS * CC * CC,         // w_sc[li]
            species, h_out);
        float* t = h_in; h_in = h_out; h_out = t;      // final h ends in h_a
    }

    hipMemsetAsync(gsum, 0, GG * sizeof(float), stream);
    hipMemsetAsync(gcnt, 0, GG * sizeof(float), stream);
    k_readout<<<(NN + 255) / 256, 256, 0, stream>>>(h_in, w_readout, ln_g, ln_b,
                                                    w_h1, b_h1, w_h2, b_h2,
                                                    graph_id, gsum, gcnt);
    k_finalize<<<1, 64, 0, stream>>>(gsum, gcnt, scale, shift, out);
    (void)in_sizes; (void)n_in; (void)out_size; (void)ws_size;
}
